// AAttnKAN_56959856279626
// MI455X (gfx1250) — matrix operand order repaired and checked by reading, not compiled, not run on hardware
//
#include <hip/hip_runtime.h>
#include <hip/hip_bf16.h>
#include <cmath>

typedef __attribute__((ext_vector_type(16))) _Float16 v16h;
typedef __attribute__((ext_vector_type(8)))  _Float16 v8h;
typedef __attribute__((ext_vector_type(8)))  float    v8f;

#define DIMC   256
#define NPIX   4096      // H*W per batch
#define QKV_OC 768
#define NA     1024      // attention sequence length per area block
#define HD     32        // head dim

__device__ __forceinline__ v16h combine16(v8h lo, v8h hi) {
    union { v16h v; v8h h[2]; } u;
    u.h[0] = lo; u.h[1] = hi;
    return u.v;
}

__device__ __forceinline__ v8f wmma_f16(v16h a, v16h b, v8f c) {
    // D = A(16x32 f16) * B(32x16 f16) + C(16x16 f32)
    return __builtin_amdgcn_wmma_f32_16x16x32_f16(false, a, false, b,
                                                  (short)0, c, false, false);
}

__device__ __forceinline__ void cvt4(v16h& t, int base, float4 f) {
    t[base + 0] = (_Float16)f.x; t[base + 1] = (_Float16)f.y;
    t[base + 2] = (_Float16)f.z; t[base + 3] = (_Float16)f.w;
}

__device__ __forceinline__ float rowmax16(float v) {
    #pragma unroll
    for (int m = 1; m < 16; m <<= 1) v = fmaxf(v, __shfl_xor(v, m, 32));
    return v;
}
__device__ __forceinline__ float rowsum16(float v) {
    #pragma unroll
    for (int m = 1; m < 16; m <<= 1) v += __shfl_xor(v, m, 32);
    return v;
}

// LDS byte offset of a generic pointer that points into LDS
__device__ __forceinline__ unsigned lds_byte_off(const void* p) {
    return (unsigned)(unsigned long long)
        (__attribute__((address_space(3))) const void*)p;
}

// CDNA5 async copy: 16 bytes global -> LDS per lane, tracked by ASYNCcnt
__device__ __forceinline__ void async_copy_b128(unsigned lds_off,
                                                const void* gaddr) {
    asm volatile("global_load_async_to_lds_b128 %0, %1, off"
                 :: "v"(lds_off), "v"((unsigned long long)gaddr)
                 : "memory");
}
__device__ __forceinline__ void wait_asynccnt0() {
    asm volatile("s_wait_asynccnt 0x0" ::: "memory");
}

// ---------------------------------------------------------------------------
// Kernel 1: qkv = W_qkv (768x256) * x (256 x N) + b, output pixel-major F16
// block = 256 threads (8 waves), tile = 128 px x 64 out, wave = 32 px x 32 out
// ---------------------------------------------------------------------------
__global__ __launch_bounds__(256) void qkv_gemm_kernel(
    const float* __restrict__ x, const float* __restrict__ w,
    const float* __restrict__ bias, _Float16* __restrict__ qkvh)
{
    __shared__ _Float16 Xs[128 * 32];   // [px][c], 8 KB, transposed-to-f16 tile

    const int b    = blockIdx.z;
    const int n0   = blockIdx.x * 128;
    const int o0   = blockIdx.y * 64;
    const int tid  = threadIdx.x;
    const int lane = tid & 31;
    const int wid  = tid >> 5;
    const int wpx  = (wid & 3) * 32;
    const int wo   = (wid >> 2) * 32;
    const int lrow = lane & 15;
    const int klo  = (lane < 16) ? 0 : 8;    // A-frag K start
    const int bklo = (lane < 16) ? 0 : 16;   // unused (B-frag K start is klo too)

    v8f acc[2][2] = {};

    for (int kc = 0; kc < DIMC; kc += 32) {
        __syncthreads();
        // stage x[b][kc..kc+31][n0..n0+127] -> Xs[px][c] (f16, transposed)
        for (int idx = tid; idx < 1024; idx += 256) {
            int cc = idx >> 5;
            int p4 = (idx & 31) << 2;
            float4 v = *reinterpret_cast<const float4*>(
                x + ((size_t)b * DIMC + kc + cc) * NPIX + n0 + p4);
            Xs[(p4 + 0) * 32 + cc] = (_Float16)v.x;
            Xs[(p4 + 1) * 32 + cc] = (_Float16)v.y;
            Xs[(p4 + 2) * 32 + cc] = (_Float16)v.z;
            Xs[(p4 + 3) * 32 + cc] = (_Float16)v.w;
        }
        __syncthreads();

        v16h afrag[2];
        #pragma unroll
        for (int s = 0; s < 2; ++s) {
            const _Float16* ab = Xs + (wpx + s * 16 + lrow) * 32 + klo;
            afrag[s] = combine16(*reinterpret_cast<const v8h*>(ab),
                                 *reinterpret_cast<const v8h*>(ab + 16));
        }
        v16h bfrag[2];
        #pragma unroll
        for (int s = 0; s < 2; ++s) {
            int oc = o0 + wo + s * 16 + lrow;
            const float* wb = w + (size_t)oc * DIMC + kc + klo;
            v16h t;
            cvt4(t,  0, *reinterpret_cast<const float4*>(wb + 0));
            cvt4(t,  4, *reinterpret_cast<const float4*>(wb + 4));
            cvt4(t,  8, *reinterpret_cast<const float4*>(wb + 16));
            cvt4(t, 12, *reinterpret_cast<const float4*>(wb + 20));
            bfrag[s] = t;
        }
        #pragma unroll
        for (int i = 0; i < 2; ++i)
            #pragma unroll
            for (int j = 0; j < 2; ++j)
                acc[i][j] = wmma_f16(afrag[i], bfrag[j], acc[i][j]);
    }

    const int hi8 = (lane < 16) ? 0 : 8;
    #pragma unroll
    for (int j = 0; j < 2; ++j) {
        int oc = o0 + wo + j * 16 + lrow;
        float bv = bias[oc];
        #pragma unroll
        for (int i = 0; i < 2; ++i) {
            #pragma unroll
            for (int r = 0; r < 8; ++r) {
                int px = n0 + wpx + i * 16 + r + hi8;
                qkvh[((size_t)b * NPIX + px) * QKV_OC + oc] =
                    (_Float16)(acc[i][j][r] + bv);
            }
        }
    }
}

// ---------------------------------------------------------------------------
// Kernel 2: flash attention, one workgroup per (ba, head).
// K (1024x32) async-copied and V^T (32x1024) staged into dynamic LDS (f16).
// Each of the 8 waves owns 16-row query tiles; online softmax; P round-trips
// through a per-wave LDS scratch to convert C-layout -> A-layout for PV.
// ---------------------------------------------------------------------------
__global__ __launch_bounds__(256) void attn_kernel(
    const _Float16* __restrict__ qkvh, float* __restrict__ attnbuf)
{
    extern __shared__ _Float16 smem[];
    _Float16* Ks = smem;                 // [j][d]  1024*32
    _Float16* Vt = Ks + NA * HD;         // [d][j]  32*1024
    _Float16* Ps = Vt + HD * NA;         // [wave][16*32]

    const int ba  = blockIdx.x;          // 0..15
    const int h   = blockIdx.y;          // 0..7
    const int b   = ba >> 2;
    const int a   = ba & 3;
    const int tid = threadIdx.x;
    const int lane = tid & 31;
    const int wid  = tid >> 5;
    const int lrow = lane & 15;
    const int klo  = (lane < 16) ? 0 : 8;
    const int bklo = (lane < 16) ? 0 : 16;
    const int hi8  = (lane < 16) ? 0 : 8;

    const size_t seq0 = (size_t)b * NPIX + a * NA;   // first pixel of block

    // ---- stage K via CDNA5 async global->LDS copy (pure 16B chunks) ----
    // 1024 rows x 64 B = 4096 chunks of 16 B
    for (int ch = tid; ch < 4096; ch += 256) {
        int j    = ch >> 2;
        int part = (ch & 3) * 8;   // halves
        const _Float16* g = qkvh + (seq0 + j) * QKV_OC + DIMC + h * HD + part;
        async_copy_b128(lds_byte_off(Ks + j * HD + part), g);
    }
    // ---- stage V^T (transpose scatter, manual) ----
    {
        const int d8 = (tid & 3) * 8;
        for (int j = tid >> 2; j < NA; j += 64) {
            const _Float16* g = qkvh + (seq0 + j) * QKV_OC + 2 * DIMC + h * HD + d8;
            v8h vv = *reinterpret_cast<const v8h*>(g);
            #pragma unroll
            for (int q = 0; q < 8; ++q) Vt[(d8 + q) * NA + j] = vv[q];
        }
    }
    wait_asynccnt0();
    __syncthreads();

    _Float16* Pw = Ps + wid * (16 * 32);
    const _Float16 hs = (_Float16)0.17677669529663687f;   // 32^-0.5

    for (int it = 0; it < 8; ++it) {
        const int i0 = (wid + it * 8) * 16;    // query tile base within Na

        // Q A-fragment (f16 direct, pre-scaled)
        v16h qa;
        {
            const _Float16* qb = qkvh + (seq0 + i0 + lrow) * QKV_OC + h * HD + klo;
            v8h qlo = *reinterpret_cast<const v8h*>(qb);
            v8h qhi = *reinterpret_cast<const v8h*>(qb + 16);
            qa = combine16(qlo * hs, qhi * hs);
        }

        float m_[8], l_[8];
        v8f o0 = {}, o1 = {};
        #pragma unroll
        for (int r = 0; r < 8; ++r) { m_[r] = -INFINITY; l_[r] = 0.f; }

        for (int jt = 0; jt < NA / 32; ++jt) {
            // K^T B-fragments for two 16-column j sub-tiles
            const _Float16* kb0 = Ks + (size_t)(jt * 32 + lrow) * HD + klo;
            const _Float16* kb1 = Ks + (size_t)(jt * 32 + 16 + lrow) * HD + klo;
            v16h bk0 = combine16(*reinterpret_cast<const v8h*>(kb0),
                                 *reinterpret_cast<const v8h*>(kb0 + 16));
            v16h bk1 = combine16(*reinterpret_cast<const v8h*>(kb1),
                                 *reinterpret_cast<const v8h*>(kb1 + 16));
            v8f z = {};
            v8f s0 = wmma_f16(qa, bk0, z);
            v8f s1 = wmma_f16(qa, bk1, z);

            // online softmax update (row stats replicated per 16-lane group)
            #pragma unroll
            for (int r = 0; r < 8; ++r) {
                float t  = rowmax16(fmaxf(s0[r], s1[r]));
                float mn = fmaxf(m_[r], t);
                float al = __expf(m_[r] - mn);
                float p0 = __expf(s0[r] - mn);
                float p1 = __expf(s1[r] - mn);
                float rs = rowsum16(p0 + p1);
                l_[r] = l_[r] * al + rs;
                m_[r] = mn;
                o0[r] *= al;
                o1[r] *= al;
                // spill P (C-layout) into per-wave row-major LDS scratch
                Pw[(r + hi8) * 32 + lrow]      = (_Float16)p0;
                Pw[(r + hi8) * 32 + 16 + lrow] = (_Float16)p1;
            }

            // reload P as A-fragment (LDS in-order within wave)
            const _Float16* pb = Pw + lrow * 32 + klo;
            v16h pa = combine16(*reinterpret_cast<const v8h*>(pb),
                                *reinterpret_cast<const v8h*>(pb + 16));

            // V B-fragments: columns d=0..15 and d=16..31
            const _Float16* vb0 = Vt + (size_t)lrow * NA + jt * 32 + klo;
            const _Float16* vb1 = Vt + (size_t)(16 + lrow) * NA + jt * 32 + klo;
            v16h bv0 = combine16(*reinterpret_cast<const v8h*>(vb0),
                                 *reinterpret_cast<const v8h*>(vb0 + 16));
            v16h bv1 = combine16(*reinterpret_cast<const v8h*>(vb1),
                                 *reinterpret_cast<const v8h*>(vb1 + 16));
            o0 = wmma_f16(pa, bv0, o0);
            o1 = wmma_f16(pa, bv1, o1);
        }

        // normalize and store (pixel-major [B][N][256], f32)
        #pragma unroll
        for (int r = 0; r < 8; ++r) {
            float inv = 1.0f / l_[r];
            size_t row = (seq0 + i0 + r + hi8) * DIMC + h * HD;
            attnbuf[row + lrow]      = o0[r] * inv;
            attnbuf[row + 16 + lrow] = o1[r] * inv;
        }
    }
}

// ---------------------------------------------------------------------------
// Kernel 3: 7x7 depthwise conv over V (read from f16 qkv buffer), += attnbuf
// ---------------------------------------------------------------------------
__global__ __launch_bounds__(256) void dwconv_kernel(
    const _Float16* __restrict__ qkvh, const float* __restrict__ wpe,
    const float* __restrict__ bpe, float* __restrict__ attnbuf)
{
    __shared__ float tile[22][22];
    __shared__ float wk[49];

    const int bc = blockIdx.z;
    const int b  = bc >> 8;
    const int c  = bc & 255;
    const int h0 = blockIdx.y * 16;
    const int w0 = blockIdx.x * 16;
    const int tx = threadIdx.x & 15;
    const int ty = threadIdx.x >> 4;

    if (threadIdx.x < 49) wk[threadIdx.x] = wpe[c * 49 + threadIdx.x];
    for (int idx = threadIdx.x; idx < 22 * 22; idx += 256) {
        int y  = idx / 22, xq = idx % 22;
        int hh = h0 + y - 3, ww = w0 + xq - 3;
        float v = 0.f;
        if (hh >= 0 && hh < 64 && ww >= 0 && ww < 64)
            v = (float)qkvh[((size_t)b * NPIX + hh * 64 + ww) * QKV_OC + 2 * DIMC + c];
        tile[y][xq] = v;
    }
    __syncthreads();

    float acc = bpe[c];
    #pragma unroll
    for (int dy = 0; dy < 7; ++dy)
        #pragma unroll
        for (int dx = 0; dx < 7; ++dx)
            acc += tile[ty + dy][tx + dx] * wk[dy * 7 + dx];

    int n = (h0 + ty) * 64 + (w0 + tx);
    attnbuf[((size_t)b * NPIX + n) * DIMC + c] += acc;
}

// ---------------------------------------------------------------------------
// Kernel 4: proj GEMM, pixel-major f32 input (A-frags straight from global),
// output scattered to NCHW d_out. block = 8 waves, 128 px x 64 out.
// ---------------------------------------------------------------------------
__global__ __launch_bounds__(256) void proj_gemm_kernel(
    const float* __restrict__ in, const float* __restrict__ w,
    const float* __restrict__ bias, float* __restrict__ out)
{
    const int b    = blockIdx.z;
    const int n0   = blockIdx.x * 128;
    const int o0   = blockIdx.y * 64;
    const int lane = threadIdx.x & 31;
    const int wid  = threadIdx.x >> 5;
    const int wpx  = (wid & 3) * 32;
    const int wo   = (wid >> 2) * 32;
    const int lrow = lane & 15;
    const int klo  = (lane < 16) ? 0 : 8;
    const int bklo = (lane < 16) ? 0 : 16;
    const int hi8  = (lane < 16) ? 0 : 8;

    v8f acc[2][2] = {};

    for (int kc = 0; kc < DIMC; kc += 32) {
        v16h afrag[2];
        #pragma unroll
        for (int s = 0; s < 2; ++s) {
            const float* ab = in + ((size_t)b * NPIX + n0 + wpx + s * 16 + lrow) * DIMC
                              + kc + klo;
            v16h t;
            cvt4(t,  0, *reinterpret_cast<const float4*>(ab + 0));
            cvt4(t,  4, *reinterpret_cast<const float4*>(ab + 4));
            cvt4(t,  8, *reinterpret_cast<const float4*>(ab + 16));
            cvt4(t, 12, *reinterpret_cast<const float4*>(ab + 20));
            afrag[s] = t;
        }
        v16h bfrag[2];
        #pragma unroll
        for (int s = 0; s < 2; ++s) {
            const float* wb = w + (size_t)(o0 + wo + s * 16 + lrow) * DIMC + kc + klo;
            v16h t;
            cvt4(t,  0, *reinterpret_cast<const float4*>(wb + 0));
            cvt4(t,  4, *reinterpret_cast<const float4*>(wb + 4));
            cvt4(t,  8, *reinterpret_cast<const float4*>(wb + 16));
            cvt4(t, 12, *reinterpret_cast<const float4*>(wb + 20));
            bfrag[s] = t;
        }
        #pragma unroll
        for (int i = 0; i < 2; ++i)
            #pragma unroll
            for (int j = 0; j < 2; ++j)
                acc[i][j] = wmma_f16(afrag[i], bfrag[j], acc[i][j]);
    }

    #pragma unroll
    for (int j = 0; j < 2; ++j) {
        int oc = o0 + wo + j * 16 + lrow;
        float bv = bias[oc];
        #pragma unroll
        for (int i = 0; i < 2; ++i) {
            #pragma unroll
            for (int r = 0; r < 8; ++r) {
                int px = n0 + wpx + i * 16 + r + hi8;
                out[((size_t)b * DIMC + oc) * NPIX + px] = acc[i][j][r] + bv;
            }
        }
    }
}

extern "C" void kernel_launch(void* const* d_in, const int* in_sizes, int n_in,
                              void* d_out, int out_size, void* d_ws, size_t ws_size,
                              hipStream_t stream) {
    const float* x      = (const float*)d_in[0];
    const float* w_qkv  = (const float*)d_in[1];
    const float* b_qkv  = (const float*)d_in[2];
    const float* w_pe   = (const float*)d_in[3];
    const float* b_pe   = (const float*)d_in[4];
    const float* w_proj = (const float*)d_in[5];
    const float* b_proj = (const float*)d_in[6];
    float* out = (float*)d_out;

    _Float16* qkvh = (_Float16*)d_ws;                    // 4*4096*768 f16 = 24 MB
    float* attnbuf = (float*)(qkvh + (size_t)4 * NPIX * QKV_OC); // 16 MB f32

    qkv_gemm_kernel<<<dim3(32, 12, 4), 256, 0, stream>>>(x, w_qkv, b_qkv, qkvh);

    const size_t attn_lds = (size_t)(NA * HD + HD * NA + 8 * 16 * 32) * sizeof(_Float16);
    attn_kernel<<<dim3(16, 8, 1), 256, attn_lds, stream>>>(qkvh, attnbuf);

    dwconv_kernel<<<dim3(4, 4, 4 * 256), 256, 0, stream>>>(qkvh, w_pe, b_pe, attnbuf);

    proj_gemm_kernel<<<dim3(32, 4, 4), 256, 0, stream>>>(attnbuf, w_proj, b_proj, out);
}
